// MultiHeadAttentionWithCache_35854386987713
// MI455X (gfx1250) — matrix operand order repaired and checked by reading, not compiled, not run on hardware
//
#include <hip/hip_runtime.h>
#include <hip/hip_bf16.h>

// ---------------------------------------------------------------------------
// MultiHeadAttentionWithCache for MI455X (gfx1250, wave32, WMMA).
// Pipeline: f32->f16 convert, QKV WMMA GEMMs, flash-attention with online
// softmax (WMMA for QK^T and P*V, permlane16 row reductions, double-buffered
// GLOBAL_LOAD_ASYNC_TO_LDS staging), output-projection WMMA GEMM.
// ---------------------------------------------------------------------------

typedef __attribute__((ext_vector_type(16))) _Float16 v16h;
typedef __attribute__((ext_vector_type(8)))  _Float16 v8h;
typedef __attribute__((ext_vector_type(8)))  float    v8f;

#define CB 2
#define CS 2048
#define CP 2048
#define CT 4096      // S + P
#define CD 1024
#define CH 16
#define CDK 64

// 1/sqrt(DK) * log2(e): softmax computed in exp2 domain.
#define SCL 0.1803368801111727f

static __device__ __forceinline__ v8f wmma_f16(v16h a, v16h b, v8f c) {
  // (neg_a, A, neg_b, B, c_mod, C, reuse_a, reuse_b)
  return __builtin_amdgcn_wmma_f32_16x16x32_f16(false, a, false, b, (short)0, c,
                                                false, false);
}

static __device__ __forceinline__ v16h join8(v8h lo, v8h hi) {
  v16h r;
#pragma unroll
  for (int i = 0; i < 8; ++i) { r[i] = lo[i]; r[8 + i] = hi[i]; }
  return r;
}

// v_permlane16_b32 based XOR butterfly within each 16-lane half
// (lane-select nibbles: lo dword = lanes 0-7, hi dword = lanes 8-15).
static __device__ __forceinline__ float perml16(float v, unsigned lo,
                                                unsigned hi) {
  int r = __builtin_amdgcn_permlane16(__float_as_int(v), __float_as_int(v), lo,
                                      hi, false, false);
  return __int_as_float(r);
}

static __device__ __forceinline__ float redmax16(float v) {
  v = fmaxf(v, perml16(v, 0x67452301u, 0xEFCDAB89u));  // xor 1
  v = fmaxf(v, perml16(v, 0x54761032u, 0xDCFE98BAu));  // xor 2
  v = fmaxf(v, perml16(v, 0x32107654u, 0xBA98FEDCu));  // xor 4
  v = fmaxf(v, perml16(v, 0xFEDCBA98u, 0x76543210u));  // xor 8
  return v;
}

static __device__ __forceinline__ float redsum16(float v) {
  v += perml16(v, 0x67452301u, 0xEFCDAB89u);
  v += perml16(v, 0x54761032u, 0xDCFE98BAu);
  v += perml16(v, 0x32107654u, 0xBA98FEDCu);
  v += perml16(v, 0xFEDCBA98u, 0x76543210u);
  return v;
}

// Async DMA: 16B per lane, global -> LDS, tracked by ASYNCcnt.
static __device__ __forceinline__ void async_b128(unsigned lds_addr,
                                                  const void* gaddr) {
  asm volatile("global_load_async_to_lds_b128 %0, %1, off" ::"v"(lds_addr),
               "v"(gaddr)
               : "memory");
}

static __device__ __forceinline__ void wait_async0() {
  asm volatile("s_wait_asynccnt 0" ::: "memory");
}

// ---------------------------------------------------------------------------
// f32 -> f16 elementwise convert (4 elements / thread, n % 1024 == 0)
// ---------------------------------------------------------------------------
__global__ __launch_bounds__(256) void k_cvt(const float* __restrict__ s,
                                             _Float16* __restrict__ d) {
  size_t i = ((size_t)blockIdx.x * 256 + threadIdx.x) * 4;
  float4 v = *(const float4*)(s + i);
  d[i + 0] = (_Float16)v.x;
  d[i + 1] = (_Float16)v.y;
  d[i + 2] = (_Float16)v.z;
  d[i + 3] = (_Float16)v.w;
}

// ---------------------------------------------------------------------------
// Scatter past_value into the t>=S region of the K / V outputs (f32, d_out)
// and into the f16 K cache [B,H,T,64] and transposed V cache [B,H,64,T].
// ---------------------------------------------------------------------------
__global__ __launch_bounds__(256) void k_scatter_past(
    const float* __restrict__ pv, float* __restrict__ kout,
    float* __restrict__ vout, _Float16* __restrict__ K_h,
    _Float16* __restrict__ Vt_h) {
  size_t i = (size_t)blockIdx.x * 256 + threadIdx.x;  // 0 .. B*H*P*DK-1
  int dk = (int)(i & 63);
  int p  = (int)((i >> 6) & (CP - 1));
  int bh = (int)(i >> 17);
  float v = pv[i];
  size_t o = ((size_t)bh * CT + CS + p) * CDK + dk;
  kout[o] = v;
  vout[o] = v;
  K_h[o]  = (_Float16)v;
  Vt_h[((size_t)bh * CDK + dk) * CT + CS + p] = (_Float16)v;
}

// ---------------------------------------------------------------------------
// WMMA GEMM: C[row,n] = sum_k A[row,k] * W[n,k]   (y = x @ W^T)
// A: [4096,1024] f16 row-major, W: [1024,1024] f16 row-major.
// 8 waves per block; wave w owns rows [blk.x*128 + w*16, +16), cols
// [blk.y*64, +64) as four 16x16 WMMA accumulators. MODE picks the epilogue:
//   0: Q -> f16 [B,H,S,64]
//   1: K -> f16 [B,H,T,64] (t<S)   + f32 d_out K section
//   2: V -> f16 [B,H,64,T] (t<S)   + f32 d_out V section (un-transposed)
//   3: O -> f32 d_out out section [B,S,D]
// ---------------------------------------------------------------------------
template <int MODE>
__global__ __launch_bounds__(256) void k_gemm(const _Float16* __restrict__ A,
                                              const _Float16* __restrict__ W,
                                              _Float16* __restrict__ out_h,
                                              float* __restrict__ out_f) {
  const int tid  = threadIdx.x;
  const int w    = tid >> 5;
  const int lane = tid & 31;
  const int nlow = lane & 15;
  const int koff = (lane >> 4) * 8;    // A-frag: hi half-wave starts at K+8
  const int kh16 = (lane >> 4) * 16;   // unused; B-frag uses koff, same k order as A
  const int row0 = blockIdx.x * 128 + w * 16;
  const int n0   = blockIdx.y * 64;

  v8f acc[4] = {};
  const _Float16* arow = A + (size_t)(row0 + nlow) * CD;

  for (int kc = 0; kc < CD; kc += 32) {
    v16h a = join8(*(const v8h*)(arow + kc + koff),
                   *(const v8h*)(arow + kc + 16 + koff));
#pragma unroll
    for (int nt = 0; nt < 4; ++nt) {
      const _Float16* wr = W + (size_t)(n0 + nt * 16 + nlow) * CD + kc + koff;
      v16h b = join8(*(const v8h*)wr, *(const v8h*)(wr + 16));
      acc[nt] = wmma_f16(a, b, acc[nt]);
    }
  }

#pragma unroll
  for (int nt = 0; nt < 4; ++nt) {
#pragma unroll
    for (int r = 0; r < 8; ++r) {
      int mr  = r + ((lane >> 4) << 3);      // C layout: hi lanes hold M+8
      int row = row0 + mr;
      int n   = n0 + nt * 16 + nlow;
      float val = acc[nt][r];
      if (MODE == 3) {
        out_f[(size_t)row * CD + n] = val;
      } else {
        int b = row >> 11;                   // row / S
        int s = row & (CS - 1);
        int h = n >> 6, dk = n & 63;
        size_t bh = (size_t)b * CH + h;
        if (MODE == 0) {
          out_h[(bh * CS + s) * CDK + dk] = (_Float16)val;
        } else if (MODE == 1) {
          size_t o = (bh * CT + s) * CDK + dk;
          out_h[o] = (_Float16)val;
          out_f[o] = val;
        } else {  // MODE == 2 (V)
          out_h[(bh * CDK + dk) * CT + s] = (_Float16)val;
          out_f[(bh * CT + s) * CDK + dk] = val;
        }
      }
    }
  }
}

// ---------------------------------------------------------------------------
// Flash attention. Grid: (S/128, B*H), 256 threads (8 waves).
// Wave w owns 16 query rows. T is consumed in 32-key chunks with
// double-buffered K [32x64] / V^T [64x32] LDS tiles filled by
// global_load_async_to_lds_b128 (one barrier per chunk):
//   - scores: 2 N-tiles x 2 K-chunks = 4 WMMAs
//   - online softmax (exp2 domain) with v_permlane16 row reductions
//   - P (f16) staged through wave-private LDS to re-layout C->A fragment
//   - output: 4 d-tiles, 1 WMMA each, rescaled accumulators
// Result written as f16 [B,S,D] for the output-projection GEMM.
// ---------------------------------------------------------------------------
__global__ __launch_bounds__(256) void k_attention(
    const _Float16* __restrict__ Qh, const _Float16* __restrict__ Kh,
    const _Float16* __restrict__ Vth, _Float16* __restrict__ attn) {
  __shared__ __align__(16) _Float16 Klds[2][32 * 64];
  __shared__ __align__(16) _Float16 Vlds[2][64 * 32];
  __shared__ __align__(16) _Float16 Plds[8][16 * 32];

  const int tid  = threadIdx.x;
  const int w    = tid >> 5;
  const int lane = tid & 31;
  const int bh   = blockIdx.y;
  const int q0   = blockIdx.x * 128 + w * 16;
  const int nlow = lane & 15;
  const int koff = (lane >> 4) * 8;
  const int kh16 = (lane >> 4) * 16;

  // Per-thread async-DMA slots: 16B of K tile + 16B of V^T tile.
  const int vr = tid >> 2, vc = (tid & 3) * 8;
  const unsigned lk[2] = {(unsigned)(uintptr_t)&Klds[0][tid * 8],
                          (unsigned)(uintptr_t)&Klds[1][tid * 8]};
  const unsigned lv[2] = {(unsigned)(uintptr_t)&Vlds[0][vr * 32 + vc],
                          (unsigned)(uintptr_t)&Vlds[1][vr * 32 + vc]};
  const _Float16* kbase = Kh + (size_t)bh * CT * CDK + tid * 8;
  const _Float16* vbase = Vth + ((size_t)bh * CDK + vr) * CT + vc;

  auto issue_tile = [&](int buf, int t0) {
    async_b128(lk[buf], kbase + (size_t)t0 * CDK);
    async_b128(lv[buf], vbase + t0);
  };

  // Q fragments for k = 0..31 and 32..63, held in registers all loop long.
  const _Float16* qrow = Qh + ((size_t)bh * CS + q0 + nlow) * CDK;
  v16h qa0 = join8(*(const v8h*)(qrow + koff), *(const v8h*)(qrow + 16 + koff));
  v16h qa1 = join8(*(const v8h*)(qrow + 32 + koff),
                   *(const v8h*)(qrow + 48 + koff));

  v8f oacc[4] = {};
  float mrun[8], lrun[8];
#pragma unroll
  for (int r = 0; r < 8; ++r) { mrun[r] = -1e30f; lrun[r] = 0.0f; }

  issue_tile(0, 0);  // prologue
  int buf = 0;

  for (int t0 = 0; t0 < CT; t0 += 32) {
    wait_async0();     // this wave's DMA into buf landed
    __syncthreads();   // all waves' DMA landed; prev buffer's readers done
    if (t0 + 32 < CT) issue_tile(buf ^ 1, t0 + 32);

    const _Float16* Kt = Klds[buf];
    const _Float16* Vt = Vlds[buf];

    // --- scores: D[16q x 16k] per N-tile, K-depth 64 via two WMMAs ---
    v8f s0 = {}, s1 = {};
#pragma unroll
    for (int kc = 0; kc < 2; ++kc) {
      const _Float16* kr0 = &Kt[(size_t)nlow * 64 + kc * 32 + koff];
      const _Float16* kr1 = &Kt[(size_t)(16 + nlow) * 64 + kc * 32 + koff];
      v16h b0 = join8(*(const v8h*)kr0, *(const v8h*)(kr0 + 16));
      v16h b1 = join8(*(const v8h*)kr1, *(const v8h*)(kr1 + 16));
      v16h aq = kc ? qa1 : qa0;
      s0 = wmma_f16(aq, b0, s0);
      s1 = wmma_f16(aq, b1, s1);
    }

    // --- online softmax (exp2 domain), permlane16 row reductions ---
    float p0[8], p1[8], corr[8];
#pragma unroll
    for (int r = 0; r < 8; ++r) {
      float a = s0[r] * SCL, b = s1[r] * SCL;
      float mx = redmax16(fmaxf(a, b));
      float mn = fmaxf(mrun[r], mx);
      corr[r] = exp2f(mrun[r] - mn);
      mrun[r] = mn;
      p0[r] = exp2f(a - mn);
      p1[r] = exp2f(b - mn);
      lrun[r] = lrun[r] * corr[r] + redsum16(p0[r] + p1[r]);
    }

    // --- P: C-layout -> LDS -> A-fragment layout (wave private) ---
#pragma unroll
    for (int r = 0; r < 8; ++r) {
      int mr = r + ((lane >> 4) << 3);
      Plds[w][mr * 32 + nlow]      = (_Float16)p0[r];
      Plds[w][mr * 32 + 16 + nlow] = (_Float16)p1[r];
    }
    asm volatile("s_wait_dscnt 0" ::: "memory");  // same-wave LDS RAW
    const _Float16* pr = &Plds[w][nlow * 32];
    v16h pa = join8(*(const v8h*)(pr + koff), *(const v8h*)(pr + 16 + koff));

    // --- rescale and accumulate O += P * V ---
#pragma unroll
    for (int dt = 0; dt < 4; ++dt) {
#pragma unroll
      for (int r = 0; r < 8; ++r) oacc[dt][r] *= corr[r];
      const _Float16* vrp = &Vt[(size_t)(dt * 16 + nlow) * 32 + koff];
      v16h vb = join8(*(const v8h*)vrp, *(const v8h*)(vrp + 16));
      oacc[dt] = wmma_f16(pa, vb, oacc[dt]);
    }
    buf ^= 1;
  }

  // epilogue: attn[b, s, h*64 + d] = O / l
  int b = bh >> 4, h = bh & 15;
#pragma unroll
  for (int r = 0; r < 8; ++r) {
    int mr = r + ((lane >> 4) << 3);
    int s  = q0 + mr;
    float inv = 1.0f / lrun[r];
#pragma unroll
    for (int dt = 0; dt < 4; ++dt) {
      attn[((size_t)b * CS + s) * CD + h * CDK + dt * 16 + nlow] =
          (_Float16)(oacc[dt][r] * inv);
    }
  }
}

// ---------------------------------------------------------------------------
extern "C" void kernel_launch(void* const* d_in, const int* in_sizes, int n_in,
                              void* d_out, int out_size, void* d_ws,
                              size_t ws_size, hipStream_t stream) {
  const float* x          = (const float*)d_in[0];
  // d_in[1] = past_key: unused (reference bug concatenates past_value twice)
  const float* past_value = (const float*)d_in[2];
  const float* Wq         = (const float*)d_in[3];
  const float* Wk         = (const float*)d_in[4];
  const float* Wv         = (const float*)d_in[5];
  const float* Wo         = (const float*)d_in[6];

  const size_t nx  = (size_t)CB * CS * CD;       // 4,194,304
  const size_t nw  = (size_t)CD * CD;            // 1,048,576
  const size_t nkv = (size_t)CB * CH * CT * CDK; // 8,388,608

  float* out  = (float*)d_out;
  float* Kout = out + nx;
  float* Vout = Kout + nkv;

  _Float16* x_h    = (_Float16*)d_ws;
  _Float16* wq_h   = x_h + nx;
  _Float16* wk_h   = wq_h + nw;
  _Float16* wv_h   = wk_h + nw;
  _Float16* wo_h   = wv_h + nw;
  _Float16* Q_h    = wo_h + nw;
  _Float16* K_h    = Q_h + nx;
  _Float16* Vt_h   = K_h + nkv;
  _Float16* attn_h = Vt_h + nkv;

  // 1) downconvert activations + weights
  k_cvt<<<dim3((unsigned)(nx / 1024)), 256, 0, stream>>>(x, x_h);
  k_cvt<<<dim3((unsigned)(nw / 1024)), 256, 0, stream>>>(Wq, wq_h);
  k_cvt<<<dim3((unsigned)(nw / 1024)), 256, 0, stream>>>(Wk, wk_h);
  k_cvt<<<dim3((unsigned)(nw / 1024)), 256, 0, stream>>>(Wv, wv_h);
  k_cvt<<<dim3((unsigned)(nw / 1024)), 256, 0, stream>>>(Wo, wo_h);

  // 2) past_value -> d_out K/V tails + f16 caches
  k_scatter_past<<<dim3((unsigned)((size_t)CB * CH * CP * CDK / 256)), 256, 0,
                   stream>>>(past_value, Kout, Vout, K_h, Vt_h);

  // 3) QKV projections
  dim3 ggrid((CB * CS) / 128, CD / 64);
  k_gemm<0><<<ggrid, 256, 0, stream>>>(x_h, wq_h, Q_h, nullptr);
  k_gemm<1><<<ggrid, 256, 0, stream>>>(x_h, wk_h, K_h, Kout);
  k_gemm<2><<<ggrid, 256, 0, stream>>>(x_h, wv_h, Vt_h, Vout);

  // 4) flash attention
  k_attention<<<dim3(CS / 128, CB * CH), 256, 0, stream>>>(Q_h, K_h, Vt_h,
                                                           attn_h);

  // 5) output projection
  k_gemm<3><<<ggrid, 256, 0, stream>>>(attn_h, wo_h, nullptr, out);
}
